// AttentionHead_68779606278470
// MI455X (gfx1250) — matrix operand order repaired and checked by reading, not compiled, not run on hardware
//
#include <hip/hip_runtime.h>
#include <hip/hip_bf16.h>
#include <math.h>

// Problem constants (from reference): X [4,4096,1024] f32, W_{Q,K,V} [64,1024] f32,
// out [4,4096,64] f32.
#define BATCH 4
#define SEQ   4096
#define EMBED 1024
#define HEAD  64
#define NTILE 4            // HEAD / 16

typedef __attribute__((ext_vector_type(16))) __bf16 v16bf;
typedef __attribute__((ext_vector_type(8)))  __bf16 v8bf;
typedef __attribute__((ext_vector_type(8)))  float  v8f;

// D = A(16x32 bf16) x B(32x16 bf16) + C(16x16 f32), f32 accumulate.
static __device__ __forceinline__ v8f wmma_bf16(v16bf a, v16bf b, v8f c) {
  return __builtin_amdgcn_wmma_f32_16x16x32_bf16(
      /*neg_a=*/false, a, /*neg_b=*/false, b,
      /*c_mod=*/(short)0, c, /*reuse_a=*/false, /*reuse_b=*/false);
}

static __device__ __forceinline__ v8f vzero8() {
  v8f z = {0.f, 0.f, 0.f, 0.f, 0.f, 0.f, 0.f, 0.f};
  return z;
}

// Reductions across the 16 lanes holding one matrix row (xor masks 1,2,4,8 stay
// inside each 16-lane half of the wave32, matching the C/D row layout).
static __device__ __forceinline__ float half16_max(float v) {
  v = fmaxf(v, __shfl_xor(v, 1, 32));
  v = fmaxf(v, __shfl_xor(v, 2, 32));
  v = fmaxf(v, __shfl_xor(v, 4, 32));
  v = fmaxf(v, __shfl_xor(v, 8, 32));
  return v;
}
static __device__ __forceinline__ float half16_sum(float v) {
  v += __shfl_xor(v, 1, 32);
  v += __shfl_xor(v, 2, 32);
  v += __shfl_xor(v, 4, 32);
  v += __shfl_xor(v, 8, 32);
  return v;
}

// ---------------------------------------------------------------------------
// Kernel 0: one-shot weight conversion f32 -> bf16 (loop-invariant data; done
// once so the projection hot loop has zero weight-side cvt VALU and half the
// weight bytes).  3 x 64 x 1024 elements = 768KB read, negligible.
// ---------------------------------------------------------------------------
__global__ void __launch_bounds__(256)
wconv_kernel(const float* __restrict__ Wq, const float* __restrict__ Wk,
             const float* __restrict__ Wv,
             __bf16* __restrict__ Wqb, __bf16* __restrict__ Wkb,
             __bf16* __restrict__ Wvb) {
  const int n = HEAD * EMBED;               // 65536 per matrix
  for (int i = blockIdx.x * blockDim.x + threadIdx.x; i < n;
       i += gridDim.x * blockDim.x) {
    Wqb[i] = (__bf16)Wq[i];
    Wkb[i] = (__bf16)Wk[i];
    Wvb[i] = (__bf16)Wv[i];
  }
}

// ---------------------------------------------------------------------------
// Kernel 1: fused QKV projection.  One wave = one 16-row tile of the flattened
// [BATCH*SEQ, EMBED] input.  X is read exactly once (the HBM-bound term of the
// whole problem); Q (pre-scaled by 1/8), K stored bf16 row-major [B*S, H];
// V stored bf16 transposed [B, H, S] so the attention kernel's P@V B-operand
// is a contiguous per-lane load.
// A-operand layout (contiguous, matching B): lane<16 holds row (lane&15),
// elems 0..15 -> K 0..15; lanes>=16 hold K 16..31.
// B-operand layout: lane n<16 holds column n, K 0..15; lane n+16 holds K 16..31.
// B-operands for one matrix are loaded as a batch into distinct registers so
// the loads clause together and WMMAs wait on partial load counts instead of
// a full-latency s_wait_loadcnt 0 per WMMA.
// ---------------------------------------------------------------------------
__global__ void __launch_bounds__(256)
qkv_proj_kernel(const float* __restrict__ X,
                const __bf16* __restrict__ Wqb,
                const __bf16* __restrict__ Wkb,
                const __bf16* __restrict__ Wvb,
                __bf16* __restrict__ Qb,
                __bf16* __restrict__ Kb,
                __bf16* __restrict__ VTb) {
  const int lane  = threadIdx.x & 31;
  const int wave  = threadIdx.x >> 5;
  const int tile  = blockIdx.x * 8 + wave;   // 1024 tiles of 16 rows
  const int row0  = tile * 16;
  const int lhalf = lane >> 4;
  const int l15   = lane & 15;
  const int base0 = lhalf * 8;               // unused; A loads use kb16 like B
  const int kb16  = lhalf * 16;              // B k-base within a 32-chunk

  v8f accQ[NTILE], accK[NTILE], accV[NTILE];
#pragma unroll
  for (int t = 0; t < NTILE; ++t) { accQ[t] = vzero8(); accK[t] = vzero8(); accV[t] = vzero8(); }

  const float* __restrict__ xrow = X + (size_t)(row0 + l15) * EMBED;

  for (int ec = 0; ec < EMBED; ec += 32) {
    // Stream the next X chunk through the prefetcher (global_prefetch_b8).
    __builtin_prefetch(xrow + ec + 64, 0, 0);

    // A-operand: rows of X, f32 -> bf16 (the only conversions in the loop).
    float4 a0 = *(const float4*)(xrow + ec + kb16);
    float4 a1 = *(const float4*)(xrow + ec + kb16 + 4);
    float4 a2 = *(const float4*)(xrow + ec + kb16 + 8);
    float4 a3 = *(const float4*)(xrow + ec + kb16 + 12);
    v16bf a;
    a[0]  = (__bf16)a0.x; a[1]  = (__bf16)a0.y; a[2]  = (__bf16)a0.z; a[3]  = (__bf16)a0.w;
    a[4]  = (__bf16)a1.x; a[5]  = (__bf16)a1.y; a[6]  = (__bf16)a1.z; a[7]  = (__bf16)a1.w;
    a[8]  = (__bf16)a2.x; a[9]  = (__bf16)a2.y; a[10] = (__bf16)a2.z; a[11] = (__bf16)a2.w;
    a[12] = (__bf16)a3.x; a[13] = (__bf16)a3.y; a[14] = (__bf16)a3.z; a[15] = (__bf16)a3.w;

    const size_t wbase = (size_t)l15 * EMBED + ec + kb16;

    // Q: batch the 4 B loads, then 4 WMMAs.
    {
      v16bf b[NTILE];
#pragma unroll
      for (int ht = 0; ht < NTILE; ++ht)
        b[ht] = *(const v16bf*)(Wqb + wbase + (size_t)ht * 16 * EMBED);
#pragma unroll
      for (int ht = 0; ht < NTILE; ++ht) accQ[ht] = wmma_bf16(a, b[ht], accQ[ht]);
    }
    // K
    {
      v16bf b[NTILE];
#pragma unroll
      for (int ht = 0; ht < NTILE; ++ht)
        b[ht] = *(const v16bf*)(Wkb + wbase + (size_t)ht * 16 * EMBED);
#pragma unroll
      for (int ht = 0; ht < NTILE; ++ht) accK[ht] = wmma_bf16(a, b[ht], accK[ht]);
    }
    // V
    {
      v16bf b[NTILE];
#pragma unroll
      for (int ht = 0; ht < NTILE; ++ht)
        b[ht] = *(const v16bf*)(Wvb + wbase + (size_t)ht * 16 * EMBED);
#pragma unroll
      for (int ht = 0; ht < NTILE; ++ht) accV[ht] = wmma_bf16(a, b[ht], accV[ht]);
    }
  }

  // C/D layout: VGPR r, lane -> (row = r + 8*lhalf, col = l15).
  const float qscale = 0.125f;  // 1/sqrt(HEAD), folded into Q
#pragma unroll
  for (int ht = 0; ht < NTILE; ++ht) {
    const int h = ht * 16 + l15;
#pragma unroll
    for (int r = 0; r < 8; ++r) {
      const int gr = row0 + r + lhalf * 8;   // flattened [B*S] row
      const int b  = gr >> 12;               // / SEQ
      const int s  = gr & (SEQ - 1);
      Qb[(size_t)gr * HEAD + h] = (__bf16)(accQ[ht][r] * qscale);
      Kb[(size_t)gr * HEAD + h] = (__bf16)(accK[ht][r]);
      VTb[((size_t)b * HEAD + h) * SEQ + s] = (__bf16)(accV[ht][r]);
    }
  }
}

// ---------------------------------------------------------------------------
// Kernel 2: causal flash attention, one wave per 16-query tile, online softmax
// over 32-key blocks.  Never materializes the 256MB score matrix.
// V-operand loads are hoisted above the softmax so their global latency hides
// behind the exp/shuffle VALU work and the LDS layout round-trip.
// ---------------------------------------------------------------------------
static __device__ void attn_qtile(const __bf16* __restrict__ Qb,
                                  const __bf16* __restrict__ Kb,
                                  const __bf16* __restrict__ VTb,
                                  float* __restrict__ Out,
                                  __bf16* __restrict__ plds,  // wave-private [16][32] bf16
                                  int batch, int t, int lane) {
  const int lhalf = lane >> 4;
  const int l15   = lane & 15;
  const int base0 = lhalf * 8;
  const int kb16  = lhalf * 16;
  const int qbase = t * 16;
  const size_t srow = (size_t)batch * SEQ;

  // Q tile in A-layout: two K=32 chunks over HEAD=64 (bf16, pre-scaled).
  const __bf16* qrow = Qb + (srow + qbase + l15) * HEAD;
  v8bf q00 = *(const v8bf*)(qrow + kb16);
  v8bf q01 = *(const v8bf*)(qrow + kb16 + 8);
  v8bf q10 = *(const v8bf*)(qrow + 32 + kb16);
  v8bf q11 = *(const v8bf*)(qrow + 32 + kb16 + 8);
  v16bf qa0 = __builtin_shufflevector(q00, q01, 0,1,2,3,4,5,6,7,8,9,10,11,12,13,14,15);
  v16bf qa1 = __builtin_shufflevector(q10, q11, 0,1,2,3,4,5,6,7,8,9,10,11,12,13,14,15);

  float m[8], l[8];
  v8f O[NTILE];
#pragma unroll
  for (int r = 0; r < 8; ++r) { m[r] = -__builtin_inff(); l[r] = 0.f; }
#pragma unroll
  for (int ht = 0; ht < NTILE; ++ht) O[ht] = vzero8();

  const int nblk = (t + 2) >> 1;             // ceil((t+1) 16-tiles / 2), 32 keys per block
  for (int jb = 0; jb < nblk; ++jb) {
    const int kbkey = jb * 32;

    // --- scores: Q(16x64) @ K^T(64x32) -> two 16x16 f32 tiles ---
    // Batch the 4 K-operand loads, then issue the 4 WMMAs.
    const __bf16* krow1 = Kb + (srow + kbkey + l15) * HEAD + kb16;  // keys kbkey..+15
    const __bf16* krow2 = krow1 + 16 * HEAD;                        // keys kbkey+16..+31
    v16bf kb0 = *(const v16bf*)(krow1);
    v16bf kb1 = *(const v16bf*)(krow1 + 32);
    v16bf kb2 = *(const v16bf*)(krow2);
    v16bf kb3 = *(const v16bf*)(krow2 + 32);

    // Hoist V-operand loads: independent of scores, consumed after softmax.
    const __bf16* vbase = VTb + ((size_t)batch * HEAD + l15) * SEQ + kbkey + kb16;
    v16bf vb[NTILE];
#pragma unroll
    for (int ht = 0; ht < NTILE; ++ht)
      vb[ht] = *(const v16bf*)(vbase + (size_t)ht * 16 * SEQ);

    v8f s1 = vzero8(), s2 = vzero8();
    s1 = wmma_bf16(qa0, kb0, s1);        // h chunk 0..31
    s1 = wmma_bf16(qa1, kb1, s1);        // h chunk 32..63
    s2 = wmma_bf16(qa0, kb2, s2);
    s2 = wmma_bf16(qa1, kb3, s2);

    // --- causal mask + online softmax (per matrix row = per accumulator VGPR) ---
    float p1[8], p2[8], fac[8];
#pragma unroll
    for (int r = 0; r < 8; ++r) {
      const int qr = qbase + r + lhalf * 8;
      const int c1 = kbkey + l15;
      const int c2 = c1 + 16;
      float a = (c1 <= qr) ? s1[r] : -__builtin_inff();
      float b = (c2 <= qr) ? s2[r] : -__builtin_inff();
      const float mt = half16_max(fmaxf(a, b));   // diag guarantees mt finite
      const float mn = fmaxf(m[r], mt);
      const float f  = __expf(m[r] - mn);
      const float e1 = __expf(a - mn);
      const float e2 = __expf(b - mn);
      m[r]   = mn;
      l[r]   = l[r] * f + half16_sum(e1 + e2);
      fac[r] = f;
      p1[r]  = e1;
      p2[r]  = e2;
    }
#pragma unroll
    for (int ht = 0; ht < NTILE; ++ht)
#pragma unroll
      for (int r = 0; r < 8; ++r) O[ht][r] *= fac[r];

    // --- C-layout -> A-layout for P via wave-private LDS (bf16 [16][32]) ---
#pragma unroll
    for (int r = 0; r < 8; ++r) {
      const int row = r + lhalf * 8;
      plds[row * 32 + l15]      = (__bf16)p1[r];
      plds[row * 32 + 16 + l15] = (__bf16)p2[r];
    }
    // Same-wave LDS ops retire in order; wait so cross-lane reads see the stores.
    asm volatile("s_wait_dscnt 0x0" ::: "memory");
    const __bf16* prow = plds + l15 * 32;
    v8bf pa0 = *(const v8bf*)(prow + kb16);
    v8bf pa1 = *(const v8bf*)(prow + kb16 + 8);
    v16bf pa = __builtin_shufflevector(pa0, pa1, 0,1,2,3,4,5,6,7,8,9,10,11,12,13,14,15);

    // --- O += P(16x32) @ V(32x64): B-operands already in registers ---
#pragma unroll
    for (int ht = 0; ht < NTILE; ++ht)
      O[ht] = wmma_bf16(pa, vb[ht], O[ht]);
  }

  // --- normalize and write out (f32, [B,S,H]) ---
#pragma unroll
  for (int r = 0; r < 8; ++r) {
    const float inv = 1.0f / l[r];
    const int gr = qbase + r + lhalf * 8;
    float* orow = Out + (srow + gr) * HEAD;
#pragma unroll
    for (int ht = 0; ht < NTILE; ++ht) orow[ht * 16 + l15] = O[ht][r] * inv;
  }
}

__global__ void __launch_bounds__(256)
flash_attn_kernel(const __bf16* __restrict__ Qb,
                  const __bf16* __restrict__ Kb,
                  const __bf16* __restrict__ VTb,
                  float* __restrict__ Out) {
  __shared__ __bf16 Plds[8][16 * 32];        // one 1KB staging tile per wave
  const int lane = threadIdx.x & 31;
  const int wave = threadIdx.x >> 5;
  const int w    = blockIdx.x * 8 + wave;    // 512 waves total
  const int batch = w >> 7;
  const int k7    = w & 127;
  // Pair tiles t and 255-t: every wave does exactly 257 key-tiles -> balanced.
  attn_qtile(Qb, Kb, VTb, Out, Plds[wave], batch, k7, lane);
  attn_qtile(Qb, Kb, VTb, Out, Plds[wave], batch, 255 - k7, lane);
}

// ---------------------------------------------------------------------------
// Launch: inputs in setup_inputs() order: embeddings, W_Q, W_K, W_V (all f32).
// Workspace: Q,K bf16 [B*S,H] + V^T bf16 [B,H,S] (6MB) + bf16 weights (384KB).
// ---------------------------------------------------------------------------
extern "C" void kernel_launch(void* const* d_in, const int* in_sizes, int n_in,
                              void* d_out, int out_size, void* d_ws, size_t ws_size,
                              hipStream_t stream) {
  const float* X  = (const float*)d_in[0];
  const float* Wq = (const float*)d_in[1];
  const float* Wk = (const float*)d_in[2];
  const float* Wv = (const float*)d_in[3];
  float* Out = (float*)d_out;

  const size_t elems = (size_t)BATCH * SEQ * HEAD;   // 1,048,576
  const size_t welems = (size_t)HEAD * EMBED;        // 65,536
  __bf16* Qb  = (__bf16*)d_ws;
  __bf16* Kb  = Qb + elems;
  __bf16* VTb = Kb + elems;
  __bf16* Wqb = VTb + elems;
  __bf16* Wkb = Wqb + welems;
  __bf16* Wvb = Wkb + welems;

  // One-shot f32->bf16 weight conversion (tiny).
  wconv_kernel<<<64, 256, 0, stream>>>(Wq, Wk, Wv, Wqb, Wkb, Wvb);
  // 1024 row-tiles, 8 waves (256 threads) per block.
  qkv_proj_kernel<<<128, 256, 0, stream>>>(X, Wqb, Wkb, Wvb, Qb, Kb, VTb);
  // 512 waves: 4 batches x 128 balanced tile-pairs.
  flash_attn_kernel<<<64, 256, 0, stream>>>(Qb, Kb, VTb, Out);
}
